// ExGRU_9543417331957
// MI455X (gfx1250) — compile-verified, matrix operand order checked by reading, not run on hardware
//
#include <hip/hip_runtime.h>
#include <math.h>

typedef unsigned short u16;
typedef __attribute__((ext_vector_type(8)))  float   v8f;
typedef __attribute__((ext_vector_type(8)))  __bf16  v8bf;
typedef __attribute__((ext_vector_type(16))) __bf16  v16bf;
typedef int v4i_t __attribute__((vector_size(16)));   // matches builtin param type

union Frag { v16bf v; v8bf h[2]; };

#define GRU_B   8192
#define GRU_IN  1024
#define GRU_H   1024
#define GRU_OUT 512
#define GRU_K1  2048   // IN + H

// ---- gfx1250 async global->LDS copy (ASYNCcnt-tracked DMA), with fallback ----
#if defined(__has_builtin)
#if __has_builtin(__builtin_amdgcn_global_load_async_to_lds_b128)
#define USE_ASYNC_COPY 1
#endif
#endif
#ifndef USE_ASYNC_COPY
#define USE_ASYNC_COPY 0
#endif

__device__ __forceinline__ void async_cp16(const u16* g, u16* l) {
#if USE_ASYNC_COPY
    __builtin_amdgcn_global_load_async_to_lds_b128(
        (__attribute__((address_space(1))) v4i_t*)g,
        (__attribute__((address_space(3))) v4i_t*)l, 0, 0);
#endif
}

__device__ __forceinline__ void wait_async() {
#if defined(__has_builtin)
#if __has_builtin(__builtin_amdgcn_s_wait_asynccnt)
    __builtin_amdgcn_s_wait_asynccnt(0);
#define HAVE_WAIT_ASYNC 1
#endif
#endif
#ifndef HAVE_WAIT_ASYNC
    asm volatile("s_wait_asynccnt 0" ::: "memory");
#endif
}

__device__ __forceinline__ u16 f2bf(float f) {
    unsigned int u = __float_as_uint(f);
    unsigned int r = u + 0x7FFFu + ((u >> 16) & 1u);   // round-to-nearest-even
    return (u16)(r >> 16);
}

// ---------------------------------------------------------------------------
// Pack bf16 concat(x, h) -> xh ; also pre-write x half of xrh.
// ---------------------------------------------------------------------------
__global__ void gru_pack(const float* __restrict__ x, const float* __restrict__ h,
                         u16* __restrict__ xh, u16* __restrict__ xrh) {
    size_t i = (size_t)blockIdx.x * blockDim.x + threadIdx.x;   // over B*2048
    int    col = (int)(i & (GRU_K1 - 1));
    size_t row = i >> 11;
    float v = (col < GRU_IN) ? x[row * GRU_IN + col]
                             : h[row * GRU_H + (col - GRU_IN)];
    u16 b = f2bf(v);
    xh[i] = b;
    if (col < GRU_IN) xrh[i] = b;
}

__global__ void gru_cvt(const float* __restrict__ src, u16* __restrict__ dst, int n) {
    int i = blockIdx.x * blockDim.x + threadIdx.x;
    if (i < n) dst[i] = f2bf(src[i]);
}

// ---------------------------------------------------------------------------
// Fused WMMA GEMM:  C[M=8192, N] = act(A[M,K] * W[N,K]^T + bias)  + GRU epilogue
//   mode 0: z = sigmoid -> zbuf
//   mode 1: r = sigmoid; xrh[:, IN+col] = bf16(r * h)
//   mode 2: ht = tanh;   h' = (1-z)h + z*ht -> hiddenOut (f32) + hbf (bf16)
//   mode 3: out = val -> outBuf
// Tile: 128x128 per block, K-step 32 (bf16), 8 wave32s x (2x4) 16x16 WMMA tiles.
// Staging: GLOBAL_LOAD_ASYNC_TO_LDS_B128 (ASYNCcnt) double-buffered.
// ---------------------------------------------------------------------------
__global__ __launch_bounds__(256) void gru_gemm(
    const u16* __restrict__ A, const u16* __restrict__ W,
    const float* __restrict__ bias, int K, int mode,
    const float* __restrict__ hstate, float* __restrict__ zbuf,
    u16* __restrict__ xrh, float* __restrict__ hiddenOut,
    u16* __restrict__ hbf, float* __restrict__ outBuf) {

    constexpr int TM = 128, TN = 128, TK = 32, LDT = 40;  // 40 shorts/row: no LDS bank conflicts
    __shared__ u16 sA[2][TM * LDT];
    __shared__ u16 sB[2][TN * LDT];

    const int t     = threadIdx.x;
    const int wave  = t >> 5;
    const int lane  = t & 31;
    const int hlf   = lane >> 4;      // 0: K-chunks {0..7,16..23}; 1: {8..15,24..31}
    const int r16   = lane & 15;
    const int waveM = wave & 3;       // 4 waves along M (4*32 = 128)
    const int waveN = wave >> 2;      // 2 waves along N (2*64 = 128)

    const size_t mBase = (size_t)blockIdx.y * TM;
    const size_t nBase = (size_t)blockIdx.x * TN;

    // --- global->LDS staging: 512 16B chunks per tile, 2 per thread ---
    const int rowS = t >> 2;            // 0..63 (second chunk adds +64 rows)
    const int ccS  = (t & 3) * 8;       // element offset within 32-wide K slice
    const u16* gA0 = A + (mBase + rowS)      * (size_t)K + ccS;
    const u16* gA1 = A + (mBase + rowS + 64) * (size_t)K + ccS;
    const u16* gB0 = W + (nBase + rowS)      * (size_t)K + ccS;
    const u16* gB1 = W + (nBase + rowS + 64) * (size_t)K + ccS;

    u16* lA0 = &sA[0][ rowS       * LDT + ccS];
    u16* lA1 = &sA[0][(rowS + 64) * LDT + ccS];
    u16* lB0 = &sB[0][ rowS       * LDT + ccS];
    u16* lB1 = &sB[0][(rowS + 64) * LDT + ccS];
    constexpr int BUFO = TM * LDT;      // element offset between buffers

    // preload K-tile 0 into buffer 0
#if USE_ASYNC_COPY
    async_cp16(gA0, lA0);
    async_cp16(gA1, lA1);
    async_cp16(gB0, lB0);
    async_cp16(gB1, lB1);
    wait_async();
#else
    {
        uint4 a0 = *(const uint4*)gA0;
        uint4 a1 = *(const uint4*)gA1;
        uint4 b0 = *(const uint4*)gB0;
        uint4 b1 = *(const uint4*)gB1;
        *(uint4*)lA0 = a0;
        *(uint4*)lA1 = a1;
        *(uint4*)lB0 = b0;
        *(uint4*)lB1 = b1;
    }
#endif
    __syncthreads();

    v8f acc[2][4];
#pragma unroll
    for (int mi = 0; mi < 2; ++mi)
#pragma unroll
        for (int ni = 0; ni < 4; ++ni) acc[mi][ni] = (v8f)0.0f;

    const int KT = K / TK;
    for (int kt = 0; kt < KT; ++kt) {
        const int  cur = kt & 1, nxt = cur ^ 1;
        const bool hasNext = (kt + 1) < KT;

#if USE_ASYNC_COPY
        if (hasNext) {   // DMA next K-tile into the other buffer; overlaps with WMMA below
            const size_t ko = (size_t)(kt + 1) * TK;
            async_cp16(gA0 + ko, lA0 + nxt * BUFO);
            async_cp16(gA1 + ko, lA1 + nxt * BUFO);
            async_cp16(gB0 + ko, lB0 + nxt * BUFO);
            async_cp16(gB1 + ko, lB1 + nxt * BUFO);
            if (kt + 2 < KT) {   // global_prefetch_b8 for K-tile kt+2
                __builtin_prefetch(gA0 + ko + TK, 0, 1);
                __builtin_prefetch(gB0 + ko + TK, 0, 1);
            }
        }
#else
        uint4 a0, a1, b0, b1;
        if (hasNext) {
            const size_t ko = (size_t)(kt + 1) * TK;
            a0 = *(const uint4*)(gA0 + ko);
            a1 = *(const uint4*)(gA1 + ko);
            b0 = *(const uint4*)(gB0 + ko);
            b1 = *(const uint4*)(gB1 + ko);
            if (kt + 2 < KT) {
                __builtin_prefetch(gA0 + ko + TK, 0, 1);
                __builtin_prefetch(gB0 + ko + TK, 0, 1);
            }
        }
#endif

        // --- fragments from LDS (layout per CDNA5 ISA 16-bit A/B 16x32) ---
        Frag af[2], bfr[4];
#pragma unroll
        for (int mi = 0; mi < 2; ++mi) {
            const int row = waveM * 32 + mi * 16 + r16;
            af[mi].h[0] = *(const v8bf*)&sA[cur][row * LDT + 8 * hlf];
            af[mi].h[1] = *(const v8bf*)&sA[cur][row * LDT + 16 + 8 * hlf];
        }
#pragma unroll
        for (int ni = 0; ni < 4; ++ni) {
            const int row = waveN * 64 + ni * 16 + r16;
            bfr[ni].h[0] = *(const v8bf*)&sB[cur][row * LDT + 8 * hlf];
            bfr[ni].h[1] = *(const v8bf*)&sB[cur][row * LDT + 16 + 8 * hlf];
        }
#pragma unroll
        for (int mi = 0; mi < 2; ++mi)
#pragma unroll
            for (int ni = 0; ni < 4; ++ni)
                acc[mi][ni] = __builtin_amdgcn_wmma_f32_16x16x32_bf16(
                    false, af[mi].v, false, bfr[ni].v,
                    (short)0, acc[mi][ni], false, false);

#if USE_ASYNC_COPY
        if (hasNext) wait_async();   // next buffer resident before the barrier
#else
        if (hasNext) {
            *(uint4*)(lA0 + nxt * BUFO) = a0;
            *(uint4*)(lA1 + nxt * BUFO) = a1;
            *(uint4*)(lB0 + nxt * BUFO) = b0;
            *(uint4*)(lB1 + nxt * BUFO) = b1;
        }
#endif
        __syncthreads();
    }

    // --- fused GRU epilogue (C/D layout: VGPR v -> M = v + 8*hlf, lane -> N = r16) ---
#pragma unroll
    for (int mi = 0; mi < 2; ++mi) {
#pragma unroll
        for (int ni = 0; ni < 4; ++ni) {
            const int col = (int)nBase + waveN * 64 + ni * 16 + r16;
            const float bv = bias[col];
#pragma unroll
            for (int v = 0; v < 8; ++v) {
                const int row = (int)mBase + waveM * 32 + mi * 16 + v + 8 * hlf;
                const float val = acc[mi][ni][v] + bv;
                const size_t idxH = (size_t)row * GRU_H + col;
                if (mode == 0) {
                    zbuf[idxH] = 1.0f / (1.0f + __expf(-val));
                } else if (mode == 1) {
                    const float r = 1.0f / (1.0f + __expf(-val));
                    xrh[(size_t)row * GRU_K1 + GRU_IN + col] = f2bf(r * hstate[idxH]);
                } else if (mode == 2) {
                    const float ht = tanhf(val);
                    const float z  = zbuf[idxH];
                    const float hn = (1.0f - z) * hstate[idxH] + z * ht;
                    hiddenOut[idxH] = hn;
                    hbf[idxH]       = f2bf(hn);
                } else {
                    outBuf[(size_t)row * GRU_OUT + col] = val;
                }
            }
        }
    }
}

// ---------------------------------------------------------------------------
extern "C" void kernel_launch(void* const* d_in, const int* in_sizes, int n_in,
                              void* d_out, int out_size, void* d_ws, size_t ws_size,
                              hipStream_t stream) {
    const float* x  = (const float*)d_in[0];
    const float* h0 = (const float*)d_in[1];
    const float* Wz = (const float*)d_in[2];
    const float* bz = (const float*)d_in[3];
    const float* Wr = (const float*)d_in[4];
    const float* br = (const float*)d_in[5];
    const float* Wh = (const float*)d_in[6];
    const float* bh = (const float*)d_in[7];
    const float* Wo = (const float*)d_in[8];
    const float* bo = (const float*)d_in[9];

    float* out       = (float*)d_out;                       // [B*OUT]
    float* hiddenOut = out + (size_t)GRU_B * GRU_OUT;       // [B*H]

    // workspace carve-out (~125 MB total)
    char*  ws  = (char*)d_ws;
    size_t off = 0;
    u16* xh_bf  = (u16*)(ws + off); off += (size_t)GRU_B * GRU_K1 * 2;   // 32 MB
    u16* xrh_bf = (u16*)(ws + off); off += (size_t)GRU_B * GRU_K1 * 2;   // 32 MB
    u16* Wz_bf  = (u16*)(ws + off); off += (size_t)GRU_H * GRU_K1 * 2;   //  4 MB
    u16* Wr_bf  = (u16*)(ws + off); off += (size_t)GRU_H * GRU_K1 * 2;   //  4 MB
    u16* Wh_bf  = (u16*)(ws + off); off += (size_t)GRU_H * GRU_K1 * 2;   //  4 MB
    u16* Wo_bf  = (u16*)(ws + off); off += (size_t)GRU_OUT * GRU_H * 2;  //  1 MB
    float* zbuf = (float*)(ws + off); off += (size_t)GRU_B * GRU_H * 4;  // 32 MB
    u16* hbf    = (u16*)(ws + off); off += (size_t)GRU_B * GRU_H * 2;    // 16 MB

    // --- precision packing ---
    {
        const size_t n = (size_t)GRU_B * GRU_K1;
        gru_pack<<<dim3((unsigned)(n / 256)), 256, 0, stream>>>(x, h0, xh_bf, xrh_bf);
        const int nw = GRU_H * GRU_K1;
        gru_cvt<<<dim3((nw + 255) / 256), 256, 0, stream>>>(Wz, Wz_bf, nw);
        gru_cvt<<<dim3((nw + 255) / 256), 256, 0, stream>>>(Wr, Wr_bf, nw);
        gru_cvt<<<dim3((nw + 255) / 256), 256, 0, stream>>>(Wh, Wh_bf, nw);
        const int no = GRU_OUT * GRU_H;
        gru_cvt<<<dim3((no + 255) / 256), 256, 0, stream>>>(Wo, Wo_bf, no);
    }

    const dim3 blk(256);
    const dim3 gridH(GRU_H / 128, GRU_B / 128);     // (8, 64)
    const dim3 gridO(GRU_OUT / 128, GRU_B / 128);   // (4, 64)

    // z = sigmoid(xh Wz^T + bz)
    gru_gemm<<<gridH, blk, 0, stream>>>(xh_bf, Wz_bf, bz, GRU_K1, 0,
                                        h0, zbuf, xrh_bf, hiddenOut, hbf, out);
    // r = sigmoid(xh Wr^T + br); xrh tail = bf16(r*h)
    gru_gemm<<<gridH, blk, 0, stream>>>(xh_bf, Wr_bf, br, GRU_K1, 1,
                                        h0, zbuf, xrh_bf, hiddenOut, hbf, out);
    // h~ = tanh(xrh Wh^T + bh); h' = (1-z)h + z h~
    gru_gemm<<<gridH, blk, 0, stream>>>(xrh_bf, Wh_bf, bh, GRU_K1, 2,
                                        h0, zbuf, xrh_bf, hiddenOut, hbf, out);
    // out = h' Wo^T + bo
    gru_gemm<<<gridO, blk, 0, stream>>>(hbf, Wo_bf, bo, GRU_H, 3,
                                        h0, zbuf, xrh_bf, hiddenOut, hbf, out);
}
